// HybridToMeModel_20074677142155
// MI455X (gfx1250) — matrix operand order repaired and checked by reading, not compiled, not run on hardware
//
#include <hip/hip_runtime.h>
#include <hip/hip_bf16.h>
#include <math.h>

#define DD   768
#define HH   12
#define DHD  64
#define BB   8
#define LL   197
#define NCAP 208          // padded tokens per image (13 * 16)
#define RC   (BB * NCAP)  // 1664 padded rows

typedef _Float16 f16;
typedef __attribute__((ext_vector_type(16))) _Float16 v16h;
typedef __attribute__((ext_vector_type(8)))  float    v8f;

// ---------------------------------------------------------------------------
// Generic GEMM: Out[m,n] = act(A[m,:K] @ W[:K,n] + bias[n]) (+ R[m,n])
// A rows live at padded stride (row index is a global padded row id).
// One wave computes a 16x64 strip: A fragment built once per 32-K step,
// then 4x v_wmma_f32_16x16x32_f16 against 4 B fragments (f32 accum).
// grid = (Nout/64, ceil(mTiles/8)), 256 threads = 8 waves.
// ---------------------------------------------------------------------------
__global__ __launch_bounds__(256)
void gemm_wmma(const float* __restrict__ A, const float* __restrict__ W,
               const float* __restrict__ bias, const float* __restrict__ R,
               float* __restrict__ Out, int K, int Nout,
               int tilesPerBatch, int mTiles, int doGelu)
{
    int lane = threadIdx.x & 31;
    int wave = threadIdx.x >> 5;
    int mt = blockIdx.y * 8 + wave;
    if (mt >= mTiles) return;                  // wave-uniform
    int bimg = mt / tilesPerBatch;
    int trow = mt - bimg * tilesPerBatch;
    int row0 = bimg * NCAP + trow * 16;
    int col0 = blockIdx.x * 64;
    int ln15 = lane & 15;
    int m = row0 + ln15;
    int n0 = col0 + ln15;
    int khalfA = (lane >> 4) << 3;             // A: +8 K offset for upper lanes
    int khalfB = (lane >> 4) << 4;             // prefetch: +16 K row offset for upper lanes

    v8f acc0 = {0.f,0.f,0.f,0.f,0.f,0.f,0.f,0.f};
    v8f acc1 = acc0, acc2 = acc0, acc3 = acc0;
    const float* Arow = A + (size_t)m * K;

    for (int kb = 0; kb < K; kb += 32) {
        if (kb + 64 < K)                       // pull next weight slab toward L2/L0
            __builtin_prefetch(&W[(size_t)(kb + 64 + khalfB) * Nout + n0], 0, 0);
        v16h a;
#pragma unroll
        for (int v = 0; v < 8; ++v) {
            int ka = (v < 4) ? (khalfA + 2 * v) : (16 + khalfA + 2 * (v - 4));
            a[2 * v]     = (f16)Arow[kb + ka];
            a[2 * v + 1] = (f16)Arow[kb + ka + 1];
        }
#pragma unroll
        for (int t = 0; t < 4; ++t) {
            int n = n0 + t * 16;
            v16h b;
#pragma unroll
            for (int v = 0; v < 8; ++v) {
                int kbk = (v < 4) ? (khalfA + 2 * v) : (16 + khalfA + 2 * (v - 4));
                b[2 * v]     = (f16)W[(size_t)(kb + kbk) * Nout + n];
                b[2 * v + 1] = (f16)W[(size_t)(kb + kbk + 1) * Nout + n];
            }
            if (t == 0)      acc0 = __builtin_amdgcn_wmma_f32_16x16x32_f16(false, a, false, b, (short)0, acc0, false, false);
            else if (t == 1) acc1 = __builtin_amdgcn_wmma_f32_16x16x32_f16(false, a, false, b, (short)0, acc1, false, false);
            else if (t == 2) acc2 = __builtin_amdgcn_wmma_f32_16x16x32_f16(false, a, false, b, (short)0, acc2, false, false);
            else             acc3 = __builtin_amdgcn_wmma_f32_16x16x32_f16(false, a, false, b, (short)0, acc3, false, false);
        }
    }

#pragma unroll
    for (int t = 0; t < 4; ++t) {
        int n = n0 + t * 16;
        v8f acc = (t == 0) ? acc0 : (t == 1) ? acc1 : (t == 2) ? acc2 : acc3;
        float bn = bias ? bias[n] : 0.f;
#pragma unroll
        for (int r = 0; r < 8; ++r) {
            int mm = row0 + ((lane < 16) ? r : 8 + r);
            float vv = acc[r] + bn;
            if (doGelu) vv = 0.5f * vv * (1.f + erff(vv * 0.70710678118654752f));
            size_t oidx = (size_t)mm * Nout + n;
            if (R) vv += R[oidx];
            Out[oidx] = vv;
        }
    }
}

// ---------------------------------------------------------------------------
// LayerNorm over D=768, one block per valid token row.
// ---------------------------------------------------------------------------
__global__ __launch_bounds__(256)
void ln_kernel(const float* __restrict__ X, float* __restrict__ Y,
               const float* __restrict__ g, const float* __restrict__ bta, int N)
{
    int row = blockIdx.x;
    int b = row / N, n = row - b * N;
    const float* x = X + (size_t)(b * NCAP + n) * DD;
    float* y = Y + (size_t)(b * NCAP + n) * DD;
    __shared__ float red[256];
    int tid = threadIdx.x;
    float s = 0.f;
    for (int d = tid; d < DD; d += 256) s += x[d];
    red[tid] = s; __syncthreads();
    for (int o = 128; o > 0; o >>= 1) { if (tid < o) red[tid] += red[tid + o]; __syncthreads(); }
    float mu = red[0] * (1.f / DD);
    __syncthreads();
    float v = 0.f;
    for (int d = tid; d < DD; d += 256) { float t = x[d] - mu; v += t * t; }
    red[tid] = v; __syncthreads();
    for (int o = 128; o > 0; o >>= 1) { if (tid < o) red[tid] += red[tid + o]; __syncthreads(); }
    float inv = rsqrtf(red[0] * (1.f / DD) + 1e-6f);
    for (int d = tid; d < DD; d += 256) y[d] = (x[d] - mu) * inv * g[d] + bta[d];
}

// ---------------------------------------------------------------------------
// Attention: one block per (b,h). K,V staged as f16 in dynamic LDS
// (2*197*64*2B + 8*197*4B = 56.7 KB). 8 waves each own queries round-robin.
// logits = q.k/8 + log(size_k); exact softmax; o = p @ V.
// ---------------------------------------------------------------------------
__global__ __launch_bounds__(256)
void attn_kernel(const float* __restrict__ qkv, const float* __restrict__ lsz,
                 float* __restrict__ out, int N)
{
    extern __shared__ char smemraw[];
    f16* Ks = (f16*)smemraw;                 // LL*DHD
    f16* Vs = Ks + LL * DHD;                 // LL*DHD
    float* Lb = (float*)(Vs + LL * DHD);     // 8*LL
    int b = blockIdx.x / HH, h = blockIdx.x % HH;
    int tid = threadIdx.x, lane = tid & 31, wave = tid >> 5;

    for (int idx = tid; idx < N * DHD; idx += 256) {
        int n = idx / DHD, d = idx - n * DHD;
        size_t base = (size_t)(b * NCAP + n) * (3 * DD) + h * DHD + d;
        Ks[idx] = (f16)qkv[base + DD];
        Vs[idx] = (f16)qkv[base + 2 * DD];
    }
    __syncthreads();

    float* lb = Lb + wave * LL;
    for (int q = wave; q < N; q += 8) {
        float qreg[DHD];
        size_t qb = (size_t)(b * NCAP + q) * (3 * DD) + h * DHD;
#pragma unroll
        for (int d = 0; d < DHD; ++d) qreg[d] = qkv[qb + d];
        float mx = -INFINITY;
        for (int k = lane; k < N; k += 32) {
            float acc = 0.f;
#pragma unroll
            for (int d = 0; d < DHD; ++d) acc += qreg[d] * (float)Ks[k * DHD + d];
            acc = acc * 0.125f + lsz[b * NCAP + k];
            lb[k] = acc;
            mx = fmaxf(mx, acc);
        }
#pragma unroll
        for (int o = 16; o > 0; o >>= 1) mx = fmaxf(mx, __shfl_xor(mx, o, 32));
        float sm = 0.f;
        for (int k = lane; k < N; k += 32) { float p = expf(lb[k] - mx); lb[k] = p; sm += p; }
#pragma unroll
        for (int o = 16; o > 0; o >>= 1) sm += __shfl_xor(sm, o, 32);
        float inv = 1.f / sm;
        for (int d = lane; d < DHD; d += 32) {
            float acc = 0.f;
            for (int k = 0; k < N; ++k) acc += lb[k] * (float)Vs[k * DHD + d];
            out[(size_t)(b * NCAP + q) * DD + h * DHD + d] = acc * inv;
        }
    }
}

// metric = mean over heads of K, then row-normalized. One block (64 thr) per token.
__global__ __launch_bounds__(64)
void metric_kernel(const float* __restrict__ qkv, float* __restrict__ mnorm, int N)
{
    int row = blockIdx.x; int b = row / N, n = row - b * N;
    int d = threadIdx.x;
    size_t base = (size_t)(b * NCAP + n) * (3 * DD) + DD + d;
    float s = 0.f;
#pragma unroll
    for (int h = 0; h < HH; ++h) s += qkv[base + h * DHD];
    s *= (1.f / HH);
    __shared__ float red[64];
    red[d] = s * s; __syncthreads();
    for (int o = 32; o > 0; o >>= 1) { if (d < o) red[d] += red[d + o]; __syncthreads(); }
    float inv = rsqrtf(red[0]);
    mnorm[(size_t)(b * NCAP + n) * DHD + d] = s * inv;
}

// Per even-token row max / argmax against odd tokens. scores[0,:] = -inf.
__global__ __launch_bounds__(128)
void nodemax_kernel(const float* __restrict__ mnorm, float* __restrict__ nmax,
                    int* __restrict__ nidx, int N)
{
    int b = blockIdx.x, i = threadIdx.x;
    int t1 = (N + 1) / 2, t2 = N / 2;
    if (i >= t1) return;
    if (i == 0) { nmax[b * 128] = -INFINITY; nidx[b * 128] = 0; return; }
    float a[DHD];
    const float* ap = mnorm + (size_t)(b * NCAP + 2 * i) * DHD;
#pragma unroll
    for (int d = 0; d < DHD; ++d) a[d] = ap[d];
    float best = -INFINITY; int bj = 0;
    for (int j = 0; j < t2; ++j) {
        const float* bp = mnorm + (size_t)(b * NCAP + 2 * j + 1) * DHD;
        float acc = 0.f;
#pragma unroll
        for (int d = 0; d < DHD; ++d) acc += a[d] * bp[d];
        if (acc > best) { best = acc; bj = j; }
    }
    nmax[b * 128 + i] = best; nidx[b * 128 + i] = bj;
}

// Stable descending ranks (== argsort(-node_max)); pos_a per reference.
__global__ __launch_bounds__(128)
void rank_kernel(const float* __restrict__ nmax, const int* __restrict__ nidx,
                 int* __restrict__ rank, int* __restrict__ posa, int N, int r)
{
    int b = blockIdx.x, i = threadIdx.x;
    int t1 = (N + 1) / 2;
    __shared__ int rs[128];
    float vi = (i < t1) ? nmax[b * 128 + i] : 0.f;
    int rk = 0;
    if (i < t1) {
        for (int j = 0; j < t1; ++j) {
            float vj = nmax[b * 128 + j];
            if (vj > vi || (vj == vi && j < i)) ++rk;
        }
    }
    rs[i] = rk;
    __syncthreads();
    if (i < t1) {
        rank[b * 128 + i] = rk;
        int p;
        if (rk < r) p = (t1 - r) + nidx[b * 128 + i];
        else { p = 0; for (int j = 0; j < i; ++j) p += (rs[j] >= r); }
        posa[b * 128 + i] = p;
    }
}

// Weighted merge: x_new = merge_sum(x*size)/size_new; updates logsize + token map.
__global__ __launch_bounds__(256)
void merge_kernel(const float* __restrict__ X, const float* __restrict__ S,
                  float* __restrict__ Xn, float* __restrict__ Sn, float* __restrict__ lsz,
                  const int* __restrict__ rank, const int* __restrict__ nidx,
                  const int* __restrict__ posa, int* __restrict__ map,
                  int mapLen, int N, int r)
{
    int b = blockIdx.x, tid = threadIdx.x;
    int t1 = (N + 1) / 2, t2 = N / 2, Nn = N - r;
    // odd ("b") tokens -> rows [t1-r, t1-r+t2)
    for (int j = 0; j < t2; ++j) {
        int src = 2 * j + 1, row = t1 - r + j;
        float s = S[b * NCAP + src];
        const float* xs = X + (size_t)(b * NCAP + src) * DD;
        float* xd = Xn + (size_t)(b * NCAP + row) * DD;
        for (int d = tid; d < DD; d += 256) xd[d] = xs[d] * s;
        if (tid == 0) Sn[b * NCAP + row] = s;
    }
    // unmerged even tokens -> rows [0, t1-r)
    for (int i = 0; i < t1; ++i) {
        if (rank[b * 128 + i] >= r) {
            int row = posa[b * 128 + i], src = 2 * i;
            float s = S[b * NCAP + src];
            const float* xs = X + (size_t)(b * NCAP + src) * DD;
            float* xd = Xn + (size_t)(b * NCAP + row) * DD;
            for (int d = tid; d < DD; d += 256) xd[d] = xs[d] * s;
            if (tid == 0) Sn[b * NCAP + row] = s;
        }
    }
    __syncthreads();
    // merged sources accumulate into destinations (serial -> deterministic)
    for (int i = 0; i < t1; ++i) {
        if (rank[b * 128 + i] < r) {
            int row = t1 - r + nidx[b * 128 + i], src = 2 * i;
            float s = S[b * NCAP + src];
            const float* xs = X + (size_t)(b * NCAP + src) * DD;
            float* xd = Xn + (size_t)(b * NCAP + row) * DD;
            for (int d = tid; d < DD; d += 256) xd[d] += xs[d] * s;
            if (tid == 0) Sn[b * NCAP + row] += s;
        }
    }
    __syncthreads();
    // normalize + log(size)
    for (int row = 0; row < Nn; ++row) {
        float s = Sn[b * NCAP + row];
        float inv = 1.f / s;
        float* xd = Xn + (size_t)(b * NCAP + row) * DD;
        for (int d = tid; d < DD; d += 256) xd[d] *= inv;
        if (tid == 0) lsz[b * NCAP + row] = logf(s);
    }
    // token map rewrite: even -> pos_a, odd -> t1-r+j
    for (int t = tid; t < mapLen; t += 256) {
        int old = map[b * mapLen + t];
        int nw = (old & 1) ? (t1 - r + (old >> 1)) : posa[b * 128 + (old >> 1)];
        map[b * mapLen + t] = nw;
    }
}

// im2col of 16x16 patches into padded token rows (token n = 1 + patch index)
__global__ __launch_bounds__(256)
void patches_kernel(const float* __restrict__ img, float* __restrict__ P)
{
    int p = blockIdx.x; int b = p / 196, pi = p - b * 196;
    int gy = pi / 14, gx = pi - gy * 14;
    float* dst = P + (size_t)(b * NCAP + 1 + pi) * DD;
    for (int col = threadIdx.x; col < DD; col += 256) {
        int c = col >> 8, r2 = col & 255, py = r2 >> 4, px = r2 & 15;
        dst[col] = img[((size_t)(b * 3 + c) * 224 + gy * 16 + py) * 224 + gx * 16 + px];
    }
}

// cls token + pos emb; init size=1, logsize=0, tok_map=arange
__global__ __launch_bounds__(256)
void embed_finish_kernel(float* __restrict__ X, const float* __restrict__ cls,
                         const float* __restrict__ pos, float* __restrict__ S,
                         float* __restrict__ lsz, int* __restrict__ tmap)
{
    int t = blockIdx.x; int b = t / LL, n = t - b * LL;
    float* x = X + (size_t)(b * NCAP + n) * DD;
    for (int d = threadIdx.x; d < DD; d += 256)
        x[d] = (n == 0 ? cls[d] : x[d]) + pos[(size_t)n * DD + d];
    if (threadIdx.x == 0) {
        S[b * NCAP + n] = 1.f;
        lsz[b * NCAP + n] = 0.f;
        tmap[b * LL + n] = n;
    }
}

__global__ void smap_init_kernel(int* smap, int N1)
{
    int idx = blockIdx.x * 256 + threadIdx.x;
    if (idx < BB * N1) smap[idx] = idx % N1;
}

// out[b,:] = xg[b, smap[b, tmap[b,0]], :] @ head_w + head_b
__global__ __launch_bounds__(128)
void head_kernel(const float* __restrict__ Xg, const int* __restrict__ tmap,
                 const int* __restrict__ smap, int N1,
                 const float* __restrict__ hw, const float* __restrict__ hb,
                 float* __restrict__ out)
{
    int idx = threadIdx.x;
    if (idx >= BB * 10) return;
    int b = idx / 10, k = idx - b * 10;
    int i0 = tmap[b * LL + 0];
    int j0 = smap[b * N1 + i0];
    const float* x = Xg + (size_t)(b * NCAP + j0) * DD;
    float acc = hb[k];
    for (int d = 0; d < DD; ++d) acc += x[d] * hw[d * 10 + k];
    out[idx] = acc;
}

// ---------------------------------------------------------------------------
extern "C" void kernel_launch(void* const* d_in, const int* in_sizes, int n_in,
                              void* d_out, int out_size, void* d_ws, size_t ws_size,
                              hipStream_t stream)
{
    (void)in_sizes; (void)n_in; (void)out_size; (void)ws_size;

    const float* img     = (const float*)d_in[0];
    const float* patch_w = (const float*)d_in[1];
    const float* patch_b = (const float*)d_in[2];
    const float* cls_tok = (const float*)d_in[3];
    const float* pos_emb = (const float*)d_in[4];
    const float* l_ng    = (const float*)d_in[5];
    const float* l_nb    = (const float*)d_in[6];
    const float* g_ng    = (const float*)d_in[7];
    const float* g_nb    = (const float*)d_in[8];
    const float* head_w  = (const float*)d_in[9];
    const float* head_b  = (const float*)d_in[10];
    const float* lp[12], *gp[12];
    for (int i = 0; i < 12; ++i) lp[i] = (const float*)d_in[11 + i];
    for (int i = 0; i < 12; ++i) gp[i] = (const float*)d_in[23 + i];
    // param order: ln1g ln1b qkv_w qkv_b proj_w proj_b ln2g ln2b fc1w fc1b fc2w fc2b

    float* ws = (float*)d_ws;
    size_t o = 0;
    auto alloc = [&](size_t n) { float* p = ws + o; o += n; return p; };
    float* xA = alloc((size_t)RC * DD);
    float* xB = alloc((size_t)RC * DD);
    float* xn = alloc((size_t)RC * DD);
    float* at = alloc((size_t)RC * DD);
    float* qk = alloc((size_t)RC * 3 * DD);
    float* h1 = alloc((size_t)RC * 4 * DD);
    float* mn = alloc((size_t)RC * DHD);
    float* sA = alloc(RC);
    float* sB = alloc(RC);
    float* lz = alloc(RC);
    float* nmax = alloc(BB * 128);
    int* nidx = (int*)alloc(BB * 128);
    int* rnk  = (int*)alloc(BB * 128);
    int* pos  = (int*)alloc(BB * 128);
    int* tmap = (int*)alloc(BB * LL);
    int* smap = (int*)alloc(BB * 160);

    auto gemm = [&](const float* Ap, const float* Wp, const float* bp, const float* Rp,
                    float* Op, int K, int Nout, int Ntok, int gelu) {
        int tpb = (Ntok + 15) / 16;
        int mt = BB * tpb;
        dim3 g(Nout / 64, (mt + 7) / 8);
        gemm_wmma<<<g, 256, 0, stream>>>(Ap, Wp, bp, Rp, Op, K, Nout, tpb, mt, gelu);
    };

    // ---- patch embedding ----
    patches_kernel<<<BB * 196, 256, 0, stream>>>(img, xn);
    gemm(xn, patch_w, patch_b, nullptr, xA, DD, DD, NCAP, 0);
    embed_finish_kernel<<<BB * LL, 256, 0, stream>>>(xA, cls_tok, pos_emb, sA, lz, tmap);

    float* xc = xA; float* xo = xB; float* sc = sA; float* so = sB;
    int N = LL;
    size_t attn_shm = (size_t)(2 * LL * DHD) * sizeof(f16) + (size_t)(8 * LL) * sizeof(float);

    auto layer = [&](const float* const* P, int li, int r, int* map, int mapLen) {
        const float* ln1g = P[0] + (size_t)li * DD;
        const float* ln1b = P[1] + (size_t)li * DD;
        const float* qw   = P[2] + (size_t)li * DD * 3 * DD;
        const float* qb   = P[3] + (size_t)li * 3 * DD;
        const float* pw   = P[4] + (size_t)li * DD * DD;
        const float* pb   = P[5] + (size_t)li * DD;
        const float* ln2g = P[6] + (size_t)li * DD;
        const float* ln2b = P[7] + (size_t)li * DD;
        const float* f1w  = P[8] + (size_t)li * DD * 4 * DD;
        const float* f1b  = P[9] + (size_t)li * 4 * DD;
        const float* f2w  = P[10] + (size_t)li * 4 * DD * DD;
        const float* f2b  = P[11] + (size_t)li * DD;

        ln_kernel<<<BB * N, 256, 0, stream>>>(xc, xn, ln1g, ln1b, N);
        gemm(xn, qw, qb, nullptr, qk, DD, 3 * DD, N, 0);
        attn_kernel<<<BB * HH, 256, attn_shm, stream>>>(qk, lz, at, N);
        metric_kernel<<<BB * N, 64, 0, stream>>>(qk, mn, N);
        gemm(at, pw, pb, xc, xc, DD, DD, N, 0);          // x += proj(attn)
        nodemax_kernel<<<BB, 128, 0, stream>>>(mn, nmax, nidx, N);
        rank_kernel<<<BB, 128, 0, stream>>>(nmax, nidx, rnk, pos, N, r);
        merge_kernel<<<BB, 256, 0, stream>>>(xc, sc, xo, so, lz, rnk, nidx, pos,
                                             map, mapLen, N, r);
        { float* t = xc; xc = xo; xo = t; t = sc; sc = so; so = t; }
        N -= r;
        ln_kernel<<<BB * N, 256, 0, stream>>>(xc, xn, ln2g, ln2b, N);
        gemm(xn, f1w, f1b, nullptr, h1, DD, 4 * DD, N, 1);   // gelu(fc1)
        gemm(h1, f2w, f2b, xc, xc, 4 * DD, DD, N, 0);        // x += fc2
    };

    for (int i = 0; i < 4; ++i)  layer(lp, i, 12, tmap, LL);   // N: 197 -> 149

    ln_kernel<<<BB * N, 256, 0, stream>>>(xc, xo, l_ng, l_nb, N);
    { float* t = xc; xc = xo; xo = t; }
    int N1 = N;                                                // 149
    smap_init_kernel<<<(BB * N1 + 255) / 256, 256, 0, stream>>>(smap, N1);

    for (int i = 0; i < 12; ++i) layer(gp, i, 8, smap, N1);    // N: 149 -> 53

    ln_kernel<<<BB * N, 256, 0, stream>>>(xc, xo, g_ng, g_nb, N);
    { float* t = xc; xc = xo; xo = t; }
    head_kernel<<<1, 128, 0, stream>>>(xc, tmap, smap, N1, head_w, head_b, (float*)d_out);
}
